// ScaledDotProductAttention_65773129171318
// MI455X (gfx1250) — matrix operand order repaired and checked by reading, not compiled, not run on hardware
//
#include <hip/hip_runtime.h>

// ---------------------------------------------------------------------------
// Complex-magnitude attention for MI455X (gfx1250, wave32, WMMA + TDM)
//   attn_real  = Qr Kr^T - Qp Kp^T      (also an output, streamed to HBM)
//   attn_phase = Qr Kp^T + Qp Kr^T
//   P = softmax(mask ? sqrt(r^2+p^2)/32 : -1e9)   (flash-style online softmax)
//   Xr = P Vr, Xp = P Vp
// Pipeline per 32-key block:
//   TDM tensor_load_to_lds (one descriptor per 32x64 f32 tile, row-padded
//   into LDS by the DMA engine; one issue per wave, TENSORcnt-tracked),
//   falling back to per-thread async-to-LDS b128, then to plain loads.
//   Cooperative f32->f16 convert: K row-major, V transposed (d-major).
//   WMMA fragments read as aligned ds_load_b128 pairs, no cvt in compute path.
// ---------------------------------------------------------------------------

typedef __attribute__((ext_vector_type(16))) _Float16 v16h;
typedef __attribute__((ext_vector_type(8)))  _Float16 h8;
typedef __attribute__((ext_vector_type(8)))  float    v8f;
typedef __attribute__((ext_vector_type(4)))  float    f32x4;
typedef int v4i __attribute__((vector_size(16)));
typedef unsigned int u32x4 __attribute__((ext_vector_type(4)));
typedef int          i32x8 __attribute__((ext_vector_type(8)));
typedef int          i32x4 __attribute__((ext_vector_type(4)));

constexpr int   B = 8, H = 16, S = 1024, D = 64;
constexpr float SCALE_INV = 1.0f / 32.0f;
constexpr float NEG_INF_V = -1000000000.0f;
constexpr float LOG2E     = 1.44269504088896340736f;

constexpr int KT         = 32;   // keys per inner block (one WMMA K-dim for PV)
constexpr int STG_STRIDE = 72;   // f32 per staging row  (64 + 8 pad)
constexpr int K16_STRIDE = 72;   // halves per K f16 row (64 + 8 pad)
constexpr int V16_STRIDE = 40;   // halves per V^T row   (32 + 8 pad)
constexpr int P_STRIDE   = 40;   // halves per p-scratch row
constexpr int NWAVE      = 4;
constexpr int WG         = NWAVE * 32;

#define HAVE_TDM       __has_builtin(__builtin_amdgcn_tensor_load_to_lds)
#define HAVE_ASYNC_LDS __has_builtin(__builtin_amdgcn_global_load_async_to_lds_b128)
#if HAVE_TDM || HAVE_ASYNC_LDS
#define STAGED 1
#else
#define STAGED 0
#endif

typedef __attribute__((address_space(1))) v4i v4i_g;   // global int4
typedef __attribute__((address_space(3))) v4i v4i_l;   // LDS int4

__device__ __forceinline__ v8f wmma16(v16h a, v16h b, v8f c) {
  return __builtin_amdgcn_wmma_f32_16x16x32_f16(false, a, false, b, (short)0, c,
                                                false, false);
}

__device__ __forceinline__ v16h combine(h8 lo, h8 hi) {
  return __builtin_shufflevector(lo, hi, 0, 1, 2, 3, 4, 5, 6, 7,
                                 8, 9, 10, 11, 12, 13, 14, 15);
}

__device__ __forceinline__ h8 cvt8(f32x4 x0, f32x4 x1) {
  h8 o;
  o[0] = (_Float16)x0[0]; o[1] = (_Float16)x0[1];
  o[2] = (_Float16)x0[2]; o[3] = (_Float16)x0[3];
  o[4] = (_Float16)x1[0]; o[5] = (_Float16)x1[1];
  o[6] = (_Float16)x1[2]; o[7] = (_Float16)x1[3];
  return o;
}

#if STAGED && !HAVE_TDM
// Async fallback: copy one 32x64 f32 tile into LDS staging (STG_STRIDE rows).
__device__ __forceinline__ void tile_copy(const float* __restrict__ g,
                                          float* l, int tid) {
#pragma unroll
  for (int i = 0; i < 4; ++i) {
    int idx = (tid + i * WG) << 2;   // f32 index in 32x64
    int row = idx >> 6;
    int col = idx & 63;
    __builtin_amdgcn_global_load_async_to_lds_b128(
        (v4i_g*)(g + row * D + col),
        (v4i_l*)(l + row * STG_STRIDE + col), 0, 0);
  }
}
#endif

__device__ __forceinline__ void wait_stage() {
#if HAVE_TDM
#if __has_builtin(__builtin_amdgcn_s_wait_tensorcnt)
  __builtin_amdgcn_s_wait_tensorcnt(0);
#else
  asm volatile("s_wait_tensorcnt 0" ::: "memory");
#endif
#elif HAVE_ASYNC_LDS
#if __has_builtin(__builtin_amdgcn_s_wait_asynccnt)
  __builtin_amdgcn_s_wait_asynccnt(0);
#else
  asm volatile("s_wait_asynccnt 0" ::: "memory");
#endif
#endif
}

__global__ __launch_bounds__(WG) void cmplx_attn_kernel(
    const float* __restrict__ q_real, const float* __restrict__ k_real,
    const float* __restrict__ v_real, const float* __restrict__ q_phase,
    const float* __restrict__ k_phase, const float* __restrict__ v_phase,
    const int* __restrict__ mask, float* __restrict__ out) {
#if STAGED
  constexpr int NBUF = 2;
  __shared__ float stage[NBUF][4][KT * STG_STRIDE];    // kr,kp,vr,vp (f32)
#else
  constexpr int NBUF = 1;
#endif
  __shared__ _Float16 kf16[NBUF][2][KT * K16_STRIDE];  // K row-major f16
  __shared__ _Float16 vT16[NBUF][2][D * V16_STRIDE];   // V transposed f16
  __shared__ _Float16 p_scr[NWAVE][16 * P_STRIDE];

  const int bh   = blockIdx.y;
  const int b    = bh / H;
  const int tid  = threadIdx.x;
  const int wave = tid >> 5;
  const int lane = tid & 31;
  const int m    = lane & 15;   // row (A) / column (B, C) index
  const int hi   = lane >> 4;   // half-wave selector
  const int q0   = blockIdx.x * (NWAVE * 16) + wave * 16;

  float* out_xr = out;
  float* out_xp = out + (size_t)B * H * S * D;
  float* out_ar = out + (size_t)2 * B * H * S * D;

  const size_t bh_off = (size_t)bh * S * D;
  const float* kr_g = k_real + bh_off;
  const float* kp_g = k_phase + bh_off;
  const float* vr_g = v_real + bh_off;
  const float* vp_g = v_phase + bh_off;

  // ---- Q fragments (A layout: lane -> row m, elem e -> K = c*32+hi*16+e)
  v16h a_qr[2], a_qp[2], a_qpn[2];
  {
    const float* qr = q_real + bh_off + (size_t)q0 * D;
    const float* qp = q_phase + bh_off + (size_t)q0 * D;
#pragma unroll
    for (int c = 0; c < 2; ++c) {
#pragma unroll
      for (int e = 0; e < 16; ++e) {
        int k = c * 32 + hi * 16 + e;
        float qv  = qr[m * D + k];
        float qpv = qp[m * D + k];
        a_qr[c][e]  = (_Float16)qv;
        a_qp[c][e]  = (_Float16)qpv;
        a_qpn[c][e] = (_Float16)(-qpv);
      }
    }
  }

  v8f acc_r[4] = {}, acc_p[4] = {};
  float mrow[8], lrow[8];
#pragma unroll
  for (int r = 0; r < 8; ++r) { mrow[r] = -3.0e38f; lrow[r] = 0.0f; }

#if HAVE_TDM
  // Each wave DMAs one of the four 32x64 f32 tiles per block via TDM.
  // D# group1: data_size=4B (code 2), pad_enable, pad every 64 DWORDs
  // (code 5) by 8 DWORDs (code 7)  ->  LDS row stride 72 f32 = STG_STRIDE.
  const float* g_src4[4] = {kr_g, kp_g, vr_g, vp_g};
  const float* my_src = g_src4[wave];
  unsigned my_lds[2];
  my_lds[0] = (unsigned)(size_t)(v4i_l*)&stage[0][wave][0];
  my_lds[1] = (unsigned)(size_t)(v4i_l*)&stage[1][wave][0];

  i32x8 dg1;
  dg1[0] = (2 << 16) | (1 << 20) | (5 << 22) | (7 << 25);
  dg1[1] = (int)(64u << 16);  // tensor_dim0 = 64   (bits 79:48)
  dg1[2] = (int)(32u << 16);  // tensor_dim1 = 32   (bits 111:80)
  dg1[3] = (int)(64u << 16);  // tile_dim0  = 64    (bits 127:112)
  dg1[4] = 32;                // tile_dim1  = 32, tile_dim2 = 0
  dg1[5] = 64;                // tensor_dim0_stride = 64 (bits 207:160)
  dg1[6] = 0;
  dg1[7] = 0;
  const i32x4 dgz4 = {};      // groups 2/3 unused (2D tensor)
  const i32x8 dgz8 = {};      // trailing descriptor group (unused)

  auto issue_block = [&](int kb, int sb) {
    unsigned long long ga =
        (unsigned long long)(size_t)(my_src + (size_t)kb * KT * D);
    u32x4 g0;
    g0[0] = 1u;                                               // count=1
    g0[1] = my_lds[sb];                                       // lds_addr
    g0[2] = (unsigned)(ga & 0xFFFFFFFFu);                     // addr[31:0]
    g0[3] = (unsigned)((ga >> 32) & 0x1FFFFFFu) | (2u << 30); // addr[56:32], type=2
    __builtin_amdgcn_tensor_load_to_lds(g0, dg1, dgz4, dgz4, dgz8, 0);
  };
#elif HAVE_ASYNC_LDS
  auto issue_block = [&](int kb, int sb) {
    const float* src[4] = {kr_g, kp_g, vr_g, vp_g};
#pragma unroll
    for (int a = 0; a < 4; ++a)
      tile_copy(src[a] + (size_t)kb * KT * D, &stage[sb][a][0], tid);
  };
#endif

  // Cooperative f32 -> f16 conversion into compact WMMA-friendly tiles.
  auto convert_block = [&](int cb, int kb) {
    (void)kb;
#if STAGED
#pragma unroll
    for (int a = 0; a < 2; ++a) {          // K arrays: row-major f16
#pragma unroll
      for (int i = 0; i < 2; ++i) {
        int idx = (tid + i * WG) * 8;
        int row = idx >> 6, col = idx & 63;
        const float* s = &stage[cb][a][row * STG_STRIDE + col];
        h8 o = cvt8(*(const f32x4*)s, *(const f32x4*)(s + 4));
        *(h8*)&kf16[cb][a][row * K16_STRIDE + col] = o;
      }
    }
#pragma unroll
    for (int a = 0; a < 2; ++a) {          // V arrays: transpose to d-major
#pragma unroll
      for (int i = 0; i < 2; ++i) {
        int idx = (tid + i * WG) * 8;
        int key = idx >> 6, d0 = idx & 63;
        const float* s = &stage[cb][2 + a][key * STG_STRIDE + d0];
        h8 o = cvt8(*(const f32x4*)s, *(const f32x4*)(s + 4));
#pragma unroll
        for (int j = 0; j < 8; ++j)
          vT16[cb][a][(d0 + j) * V16_STRIDE + key] = o[j];
      }
    }
#else
    const float* gk[2] = {kr_g, kp_g};
    const float* gv[2] = {vr_g, vp_g};
#pragma unroll
    for (int a = 0; a < 2; ++a) {
#pragma unroll
      for (int i = 0; i < 2; ++i) {
        int idx = (tid + i * WG) * 8;
        int row = idx >> 6, col = idx & 63;
        const float* s = gk[a] + (size_t)(kb * KT + row) * D + col;
        h8 o = cvt8(*(const f32x4*)s, *(const f32x4*)(s + 4));
        *(h8*)&kf16[cb][a][row * K16_STRIDE + col] = o;
      }
    }
#pragma unroll
    for (int a = 0; a < 2; ++a) {
#pragma unroll
      for (int i = 0; i < 2; ++i) {
        int idx = (tid + i * WG) * 8;
        int key = idx >> 6, d0 = idx & 63;
        const float* s = gv[a] + (size_t)(kb * KT + key) * D + d0;
        h8 o = cvt8(*(const f32x4*)s, *(const f32x4*)(s + 4));
#pragma unroll
        for (int j = 0; j < 8; ++j)
          vT16[cb][a][(d0 + j) * V16_STRIDE + key] = o[j];
      }
    }
#endif
  };

  constexpr int NKB = S / KT;  // 32
#if STAGED
  issue_block(0, 0);
  wait_stage();
  __syncthreads();
#endif

  for (int kb = 0; kb < NKB; ++kb) {
    const int cb = kb % NBUF;
#if STAGED
    if (kb + 1 < NKB) issue_block(kb + 1, cb ^ 1);  // overlap with convert+compute
#else
    __syncthreads();  // previous compute done before overwriting f16 tiles
#endif
    convert_block(cb, kb);
    __syncthreads();  // f16 tiles visible to all waves

    const _Float16* krl = &kf16[cb][0][0];
    const _Float16* kpl = &kf16[cb][1][0];
    const _Float16* vrl = &vT16[cb][0][0];
    const _Float16* vpl = &vT16[cb][1][0];

    // prefetch next mask line for this lane's first row (keep near: locality 3)
    __builtin_prefetch(mask + ((size_t)b * S + q0 + 8 * hi) * S + (kb + 1) * KT + m, 0, 3);

    // ---- scores: two 16x16 tiles over keys [kb*32, kb*32+32)
    v8f cr[2], cp[2];
#pragma unroll
    for (int t = 0; t < 2; ++t) {
      v8f r = {}, p = {};
#pragma unroll
      for (int c = 0; c < 2; ++c) {
        // B layout: lane -> key column (t*16+m), elem e -> d = c*32 + hi*16 + e
        int off = (t * 16 + m) * K16_STRIDE + c * 32 + hi * 16;
        v16h bkr = combine(*(const h8*)&krl[off], *(const h8*)&krl[off + 8]);
        v16h bkp = combine(*(const h8*)&kpl[off], *(const h8*)&kpl[off + 8]);
        r = wmma16(a_qr[c],  bkr, r);
        r = wmma16(a_qpn[c], bkp, r);   // (-Qp)·Kp
        p = wmma16(a_qr[c],  bkp, p);
        p = wmma16(a_qp[c],  bkr, p);
      }
      cr[t] = r; cp[t] = p;
    }

    // ---- magnitude, attn_real output stream, mask
    float sc[2][8];
#pragma unroll
    for (int t = 0; t < 2; ++t) {
      int kcol = kb * KT + t * 16 + m;
#pragma unroll
      for (int r = 0; r < 8; ++r) {
        int   q  = q0 + r + 8 * hi;
        float ar = cr[t][r], ap = cp[t][r];
        out_ar[((size_t)bh * S + q) * S + kcol] = ar;
        float amp = sqrtf(ar * ar + ap * ap) * SCALE_INV;
        int   mv  = mask[((size_t)b * S + q) * S + kcol];
        sc[t][r]  = (mv == 0) ? NEG_INF_V : amp;
      }
    }

    // ---- online softmax (rows live in 16-lane groups)
    float pv[2][8], sfac[8];
#pragma unroll
    for (int r = 0; r < 8; ++r) {
      float rm = fmaxf(sc[0][r], sc[1][r]);
#pragma unroll
      for (int off = 8; off >= 1; off >>= 1)
        rm = fmaxf(rm, __shfl_xor(rm, off, 16));
      float mnew = fmaxf(mrow[r], rm);
      float f    = exp2f((mrow[r] - mnew) * LOG2E);
      float p0   = exp2f((sc[0][r] - mnew) * LOG2E);
      float p1   = exp2f((sc[1][r] - mnew) * LOG2E);
      float rs   = p0 + p1;
#pragma unroll
      for (int off = 8; off >= 1; off >>= 1)
        rs += __shfl_xor(rs, off, 16);
      lrow[r] = lrow[r] * f + rs;
      mrow[r] = mnew;
      sfac[r] = f;
      pv[0][r] = p0; pv[1][r] = p1;
    }
#pragma unroll
    for (int t = 0; t < 4; ++t)
#pragma unroll
      for (int r = 0; r < 8; ++r) {
        acc_r[t][r] *= sfac[r];
        acc_p[t][r] *= sfac[r];
      }

    // ---- P: C-layout -> A-layout via per-wave LDS scratch
    _Float16* pl = &p_scr[wave][0];
#pragma unroll
    for (int t = 0; t < 2; ++t)
#pragma unroll
      for (int r = 0; r < 8; ++r)
        pl[(r + 8 * hi) * P_STRIDE + t * 16 + m] = (_Float16)pv[t][r];
    __builtin_amdgcn_wave_barrier();

    v16h a_pf;
    {
      int o0 = m * P_STRIDE + hi * 16;
      a_pf = combine(*(const h8*)&pl[o0], *(const h8*)&pl[o0 + 8]);
    }

    // ---- PV accumulation: 4 d-tiles x {real, phase} from transposed V
#pragma unroll
    for (int dt = 0; dt < 4; ++dt) {
      int ro = (dt * 16 + m) * V16_STRIDE + hi * 16;
      v16h bvr = combine(*(const h8*)&vrl[ro], *(const h8*)&vrl[ro + 8]);
      v16h bvp = combine(*(const h8*)&vpl[ro], *(const h8*)&vpl[ro + 8]);
      acc_r[dt] = wmma16(a_pf, bvr, acc_r[dt]);
      acc_p[dt] = wmma16(a_pf, bvp, acc_p[dt]);
    }

#if STAGED
    wait_stage();
    __syncthreads();
#endif
  }

  // ---- normalize and write x_real / x_phase
#pragma unroll
  for (int r = 0; r < 8; ++r) {
    float inv = 1.0f / lrow[r];
    int   q   = q0 + r + 8 * hi;
#pragma unroll
    for (int dt = 0; dt < 4; ++dt) {
      size_t o = ((size_t)bh * S + q) * D + dt * 16 + m;
      out_xr[o] = acc_r[dt][r] * inv;
      out_xp[o] = acc_p[dt][r] * inv;
    }
  }
}

extern "C" void kernel_launch(void* const* d_in, const int* in_sizes, int n_in,
                              void* d_out, int out_size, void* d_ws,
                              size_t ws_size, hipStream_t stream) {
  (void)in_sizes; (void)n_in; (void)out_size; (void)d_ws; (void)ws_size;
  const float* q_real  = (const float*)d_in[0];
  const float* k_real  = (const float*)d_in[1];
  const float* v_real  = (const float*)d_in[2];
  const float* q_phase = (const float*)d_in[3];
  const float* k_phase = (const float*)d_in[4];
  const float* v_phase = (const float*)d_in[5];
  const int*   mask    = (const int*)d_in[6];

  dim3 grid(S / (NWAVE * 16), B * H);  // (16, 128)
  cmplx_attn_kernel<<<grid, WG, 0, stream>>>(q_real, k_real, v_real, q_phase,
                                             k_phase, v_phase, mask,
                                             (float*)d_out);
}
